// RNNNet_53961969107394
// MI455X (gfx1250) — matrix operand order repaired and checked by reading, not compiled, not run on hardware
//
#include <hip/hip_runtime.h>

// Problem dims (fixed by the reference)
#define HDIM 512
#define TSEQ 1024
#define BTOT 256
#define BT   16              // batch rows per workgroup
#define NTW  4               // 16-wide N tiles per wave (8 waves * 4 * 16 = 512)
#define KC   16              // K chunks of 32 (512/32)

typedef __attribute__((ext_vector_type(16))) _Float16 v16h;
typedef __attribute__((ext_vector_type(8)))  _Float16 v8h;
typedef __attribute__((ext_vector_type(8)))  float    v8f;

// Global-address-space fragment pointer: keeps weight streams on
// global_load_b128 (LOADcnt only) instead of flat_load (LOADcnt+DScnt).
typedef const __attribute__((address_space(1))) v16h* gfrag_t;

#define WMMA_F16(a, b, c) __builtin_amdgcn_wmma_f32_16x16x32_f16( \
        false, (a), false, (b), (short)0, (c), false, false)

// Branchless tanh: prefer CDNA5's hardware V_TANH_F32; otherwise use the
// hardware exp2/rcp units (tanh(x) = 1 - 2/(e^{2x}+1)). Either way, no
// exec-mask divergence on the recurrence critical path.
__device__ __forceinline__ float fast_tanh(float x) {
#if __has_builtin(__builtin_amdgcn_tanhf)
    return __builtin_amdgcn_tanhf(x);
#else
    float t = __builtin_amdgcn_exp2f(x * 2.885390081777927f); // e^(2x)
    return 1.0f - 2.0f * __builtin_amdgcn_rcpf(t + 1.0f);
#endif
}

// ---------------------------------------------------------------------------
// Pack a [H,H] fp32 weight W (used as h @ W^T, i.e. B[k][n] = W[n][k]) into
// WMMA B-fragment order, f16:
//   fragment f = ntile*KC + kc   covers k in [kc*32, +32), n in [ntile*16, +16)
//   within a fragment: lane l -> n_local = l%16; element j -> k_local = 16*(j/8) + 8*(l/16) + j%8
// Each lane of the compute kernel loads its 16 halfs (32B) contiguously.
// ---------------------------------------------------------------------------
__global__ void pack_weight_f16(const float* __restrict__ W, _Float16* __restrict__ dst) {
    int idx = blockIdx.x * blockDim.x + threadIdx.x;
    if (idx >= HDIM * HDIM) return;
    int nl = (idx >> 4) & 15;   // lane%16  -> n_local
    int kl = ((idx & 8) << 1) + (((idx >> 8) & 1) << 3) + (idx & 7); // element, lane/16 -> k_local
    int f  = idx >> 9;          // fragment id
    int kc = f & (KC - 1);
    int nt = f >> 4;
    dst[idx] = (_Float16)W[(nt * 16 + nl) * HDIM + (kc * 32 + kl)];
}

__global__ void prep_vec(const float* __restrict__ W_ih1,
                         const float* __restrict__ b_ih1, const float* __restrict__ b_hh1,
                         const float* __restrict__ b_ih2, const float* __restrict__ b_hh2,
                         _Float16* __restrict__ wih1h,
                         float* __restrict__ bias1, float* __restrict__ bias2) {
    int i = blockIdx.x * blockDim.x + threadIdx.x;
    if (i < HDIM) {
        wih1h[i] = (_Float16)W_ih1[i];          // W_ih1 is [H,1]
        bias1[i] = b_ih1[i] + b_hh1[i];
        bias2[i] = b_ih2[i] + b_hh2[i];
    }
}

// ---------------------------------------------------------------------------
// Persistent 2-layer RNN kernel: one workgroup owns BT=16 batch rows for all
// T steps. h1,h2 live in LDS (f16). 8 waves; wave w owns N tiles [4w,4w+4).
// Weights are streamed from L2 every step with an explicit register
// ping-pong (load kc+1 while WMMAs of kc execute).
// ---------------------------------------------------------------------------
__global__ __launch_bounds__(256, 1)
void rnn2_wmma(const float* __restrict__ x,
               const _Float16* __restrict__ Whh1p,
               const _Float16* __restrict__ Wih2p,
               const _Float16* __restrict__ Whh2p,
               const _Float16* __restrict__ wih1,
               const float* __restrict__ bias1,
               const float* __restrict__ bias2,
               const float* __restrict__ wlin,
               const float* __restrict__ blin,
               float* __restrict__ out)
{
    __shared__ __align__(16) _Float16 h1[BT * HDIM];   // 16 KB
    __shared__ __align__(16) _Float16 h2[BT * HDIM];   // 16 KB
    __shared__ float  sb1[HDIM], sb2[HDIM], swl[HDIM]; //  6 KB
    __shared__ _Float16 swih1[HDIM];                   //  1 KB
    __shared__ float  sx[BT];
    __shared__ float  sred[256];

    const int tid  = threadIdx.x;
    const int wave = tid >> 5;
    const int lane = tid & 31;
    const int b0   = blockIdx.x * BT;

    for (int i = tid; i < HDIM; i += 256) {
        sb1[i] = bias1[i]; sb2[i] = bias2[i];
        swl[i] = wlin[i];  swih1[i] = wih1[i];
    }
    for (int i = tid; i < BT * HDIM / 2; i += 256) {   // h0 = 0
        ((unsigned int*)h1)[i] = 0u;
        ((unsigned int*)h2)[i] = 0u;
    }
    __syncthreads();

    const float bl   = blin[0];
    const int   mrow = lane & 15;             // A-fragment row (batch)
    const int   khi  = (lane >> 4) << 3;      // A-fragment K half-select (0 / 8)
    const int   moff = (lane >> 4) << 3;      // C/D M offset (0 / 8)
    const int   ncol = lane & 15;             // C/D N within tile
    // fragment element index for this lane, tile i, chunk kc:
    //   fidx(i,kc) = fbase + (i*KC + kc)*32
    const int   fbase = (wave * NTW * KC) * 32 + lane;

    for (int t = 0; t < TSEQ; ++t) {
        // Re-materialize weight bases each step so the compiler cannot hoist
        // the (t-invariant) weight loads out of the time loop and spill;
        // then restore the global address space so loads stay global_*.
        unsigned long long uw1  = (unsigned long long)Whh1p;
        unsigned long long uw2i = (unsigned long long)Wih2p;
        unsigned long long uw2h = (unsigned long long)Whh2p;
        asm volatile("" : "+s"(uw1), "+s"(uw2i), "+s"(uw2h));
        gfrag_t B1  = (gfrag_t)uw1;
        gfrag_t B2i = (gfrag_t)uw2i;
        gfrag_t B2h = (gfrag_t)uw2h;

        if (tid < BT) sx[tid] = x[(b0 + tid) * TSEQ + t];
        __syncthreads();

        // ---------- layer 1: z1 = x*wih1 + bias1 + h1 @ Whh1^T ----------
        v8f acc[NTW];
        #pragma unroll
        for (int i = 0; i < NTW; ++i) {
            int n = (wave * NTW + i) * 16 + ncol;
            float base = sb1[n];
            float w    = (float)swih1[n];
            #pragma unroll
            for (int r = 0; r < 8; ++r) acc[i][r] = base + sx[moff + r] * w;
        }
        {
            v16h bA[NTW], bB[NTW];
            #pragma unroll
            for (int i = 0; i < NTW; ++i) bA[i] = B1[fbase + (i * KC + 0) * 32];
            #pragma unroll 1
            for (int kc = 0; kc < KC; kc += 2) {
                #pragma unroll
                for (int i = 0; i < NTW; ++i) bB[i] = B1[fbase + (i * KC + kc + 1) * 32];
                union { v16h v; v8h h[2]; } a;
                const v8h* hp = (const v8h*)&h1[mrow * HDIM + kc * 32 + khi];
                a.h[0] = hp[0]; a.h[1] = hp[2];
                #pragma unroll
                for (int i = 0; i < NTW; ++i) acc[i] = WMMA_F16(a.v, bA[i], acc[i]);
                if (kc + 2 < KC) {
                    #pragma unroll
                    for (int i = 0; i < NTW; ++i) bA[i] = B1[fbase + (i * KC + kc + 2) * 32];
                }
                const v8h* hq = (const v8h*)&h1[mrow * HDIM + (kc + 1) * 32 + khi];
                a.h[0] = hq[0]; a.h[1] = hq[2];
                #pragma unroll
                for (int i = 0; i < NTW; ++i) acc[i] = WMMA_F16(a.v, bB[i], acc[i]);
            }
        }
        __syncthreads();             // everyone finished reading old h1
        #pragma unroll
        for (int i = 0; i < NTW; ++i) {
            int n = (wave * NTW + i) * 16 + ncol;
            #pragma unroll
            for (int r = 0; r < 8; ++r)
                h1[(moff + r) * HDIM + n] = (_Float16)fast_tanh(acc[i][r]);
        }
        __syncthreads();             // new h1 visible

        // ---------- layer 2: z2 = bias2 + h1 @ Wih2^T + h2 @ Whh2^T ----------
        #pragma unroll
        for (int i = 0; i < NTW; ++i) {
            float base = sb2[(wave * NTW + i) * 16 + ncol];
            #pragma unroll
            for (int r = 0; r < 8; ++r) acc[i][r] = base;
        }
        {
            v16h biA[NTW], bhA[NTW], biB[NTW], bhB[NTW];
            #pragma unroll
            for (int i = 0; i < NTW; ++i) {
                biA[i] = B2i[fbase + (i * KC + 0) * 32];
                bhA[i] = B2h[fbase + (i * KC + 0) * 32];
            }
            #pragma unroll 1
            for (int kc = 0; kc < KC; kc += 2) {
                #pragma unroll
                for (int i = 0; i < NTW; ++i) {
                    biB[i] = B2i[fbase + (i * KC + kc + 1) * 32];
                    bhB[i] = B2h[fbase + (i * KC + kc + 1) * 32];
                }
                union { v16h v; v8h h[2]; } a1, a2;
                const v8h* hp1 = (const v8h*)&h1[mrow * HDIM + kc * 32 + khi];
                const v8h* hp2 = (const v8h*)&h2[mrow * HDIM + kc * 32 + khi];
                a1.h[0] = hp1[0]; a1.h[1] = hp1[2];
                a2.h[0] = hp2[0]; a2.h[1] = hp2[2];
                #pragma unroll
                for (int i = 0; i < NTW; ++i) {
                    acc[i] = WMMA_F16(a1.v, biA[i], acc[i]);
                    acc[i] = WMMA_F16(a2.v, bhA[i], acc[i]);
                }
                if (kc + 2 < KC) {
                    #pragma unroll
                    for (int i = 0; i < NTW; ++i) {
                        biA[i] = B2i[fbase + (i * KC + kc + 2) * 32];
                        bhA[i] = B2h[fbase + (i * KC + kc + 2) * 32];
                    }
                }
                const v8h* hq1 = (const v8h*)&h1[mrow * HDIM + (kc + 1) * 32 + khi];
                const v8h* hq2 = (const v8h*)&h2[mrow * HDIM + (kc + 1) * 32 + khi];
                a1.h[0] = hq1[0]; a1.h[1] = hq1[2];
                a2.h[0] = hq2[0]; a2.h[1] = hq2[2];
                #pragma unroll
                for (int i = 0; i < NTW; ++i) {
                    acc[i] = WMMA_F16(a1.v, biB[i], acc[i]);
                    acc[i] = WMMA_F16(a2.v, bhB[i], acc[i]);
                }
            }
        }
        __syncthreads();             // everyone finished reading old h2
        #pragma unroll
        for (int i = 0; i < NTW; ++i) {
            int n = (wave * NTW + i) * 16 + ncol;
            #pragma unroll
            for (int r = 0; r < 8; ++r)
                h2[(moff + r) * HDIM + n] = (_Float16)fast_tanh(acc[i][r]);
        }
        __syncthreads();             // new h2 visible

        // ---------- output: out[b,t] = h2[b,:] . wlin + blin ----------
        {
            int b   = tid >> 4;      // 16 batch rows
            int seg = tid & 15;      // 16 segments of 32 cols
            const v8h*   hp = (const v8h*)&h2[b * HDIM + seg * 32];
            const float* wp = &swl[seg * 32];
            float p = 0.f;
            #pragma unroll
            for (int c = 0; c < 4; ++c) {
                v8h v = hp[c];
                #pragma unroll
                for (int j = 0; j < 8; ++j) p += (float)v[j] * wp[c * 8 + j];
            }
            sred[tid] = p;
            __syncthreads();
            if (seg == 0) {
                float s = bl;
                #pragma unroll
                for (int j = 0; j < 16; ++j) s += sred[(b << 4) + j];
                out[(b0 + b) * TSEQ + t] = s;
            }
            __syncthreads();         // sred / h2 safe for next step
        }
    }
}

// ---------------------------------------------------------------------------
// Host-side launcher. d_in order: input, future, W_ih1, W_hh1, b_ih1, b_hh1,
// W_ih2, W_hh2, b_ih2, b_hh2, W_lin, b_lin.
// ---------------------------------------------------------------------------
extern "C" void kernel_launch(void* const* d_in, const int* in_sizes, int n_in,
                              void* d_out, int out_size, void* d_ws, size_t ws_size,
                              hipStream_t stream) {
    const float* input = (const float*)d_in[0];
    const float* W_ih1 = (const float*)d_in[2];
    const float* W_hh1 = (const float*)d_in[3];
    const float* b_ih1 = (const float*)d_in[4];
    const float* b_hh1 = (const float*)d_in[5];
    const float* W_ih2 = (const float*)d_in[6];
    const float* W_hh2 = (const float*)d_in[7];
    const float* b_ih2 = (const float*)d_in[8];
    const float* b_hh2 = (const float*)d_in[9];
    const float* W_lin = (const float*)d_in[10];
    const float* b_lin = (const float*)d_in[11];

    // Workspace layout (bytes)
    char* ws = (char*)d_ws;
    const size_t MAT = (size_t)HDIM * HDIM * sizeof(_Float16);   // 512 KB
    _Float16* Whh1p = (_Float16*)(ws);
    _Float16* Wih2p = (_Float16*)(ws + MAT);
    _Float16* Whh2p = (_Float16*)(ws + 2 * MAT);
    _Float16* wih1h = (_Float16*)(ws + 3 * MAT);
    float*    bias1 = (float*)   (ws + 3 * MAT + HDIM * sizeof(_Float16));
    float*    bias2 = bias1 + HDIM;

    const int nelem = HDIM * HDIM;
    pack_weight_f16<<<(nelem + 255) / 256, 256, 0, stream>>>(W_hh1, Whh1p);
    pack_weight_f16<<<(nelem + 255) / 256, 256, 0, stream>>>(W_ih2, Wih2p);
    pack_weight_f16<<<(nelem + 255) / 256, 256, 0, stream>>>(W_hh2, Whh2p);
    prep_vec<<<(HDIM + 255) / 256, 256, 0, stream>>>(W_ih1, b_ih1, b_hh1, b_ih2, b_hh2,
                                                     wih1h, bias1, bias2);

    rnn2_wmma<<<BTOT / BT, 256, 0, stream>>>(input, Whh1p, Wih2p, Whh2p,
                                             wih1h, bias1, bias2,
                                             W_lin, b_lin, (float*)d_out);
}
